// PreLNTransformerBlock_7705171329093
// MI455X (gfx1250) — matrix operand order repaired and checked by reading, not compiled, not run on hardware
//
#include <hip/hip_runtime.h>
#include <stdint.h>

// ---------------------------------------------------------------------------
// Types
// ---------------------------------------------------------------------------
typedef __bf16 bf16;
typedef __attribute__((ext_vector_type(8)))  __bf16 v8bf;
typedef __attribute__((ext_vector_type(16))) __bf16 v16bf;
typedef __attribute__((ext_vector_type(8)))  float  v8f;
typedef __attribute__((ext_vector_type(4)))  float  v4f;

// ---------------------------------------------------------------------------
// bf16 <-> f32 helpers (round-to-nearest-even)
// ---------------------------------------------------------------------------
__device__ __forceinline__ bf16 f2bf(float f) {
    uint32_t u; __builtin_memcpy(&u, &f, 4);
    u += 0x7FFFu + ((u >> 16) & 1u);
    uint16_t s = (uint16_t)(u >> 16);
    bf16 b; __builtin_memcpy(&b, &s, 2);
    return b;
}
__device__ __forceinline__ float bf2f(bf16 b) {
    uint16_t s; __builtin_memcpy(&s, &b, 2);
    uint32_t u = ((uint32_t)s) << 16;
    float f; __builtin_memcpy(&f, &u, 4);
    return f;
}
__device__ __forceinline__ v16bf cat8(v8bf lo, v8bf hi) {
    return __builtin_shufflevector(lo, hi, 0,1,2,3,4,5,6,7,8,9,10,11,12,13,14,15);
}
__device__ __forceinline__ v8f wmma_bf16(v16bf a, v16bf b, v8f c) {
    // v_wmma_f32_16x16x32_bf16  D = A(16x32) * B(32x16) + C
    return __builtin_amdgcn_wmma_f32_16x16x32_bf16(false, a, false, b,
                                                   (short)0, c, false, false);
}

// ---------------------------------------------------------------------------
// CDNA5 async global->LDS copy (16B per lane), tracked with ASYNCcnt.
// Generic pointers to __shared__ carry the wave-relative LDS byte offset in
// their low 32 bits (flat LDS aperture, ISA 10.2), which is exactly what the
// VDST operand of global_load_async_to_lds_* wants.
// ---------------------------------------------------------------------------
__device__ __forceinline__ void async_copy16(void* lds_dst, const void* gsrc) {
    uint32_t loff = (uint32_t)(uintptr_t)lds_dst;
    uint64_t gaddr = (uint64_t)(uintptr_t)gsrc;
    asm volatile("global_load_async_to_lds_b128 %0, %1, off"
                 :: "v"(loff), "v"(gaddr) : "memory");
}
__device__ __forceinline__ void wait_async_le(int n) {
    if (n == 0) asm volatile("s_wait_asynccnt 0" ::: "memory");
    else        asm volatile("s_wait_asynccnt 4" ::: "memory");
}

// A fragment (16x32 bf16, MxK): lane m = lane&15 is the row; in the K order
// used here (as for B) the lane holds 16 contiguous K values starting at K = 16*half.
__device__ __forceinline__ v16bf load_afrag(const bf16* base, int stride) {
    const int lane = threadIdx.x & 31;
    const bf16* p = base + (lane & 15) * stride + 16 * (lane >> 4);
    return cat8(*(const v8bf*)p, *(const v8bf*)(p + 8));
}
// B fragment (32x16 bf16, KxN): lane n = lane&15 is the column; the lane holds
// 16 contiguous K values starting at K = 16*half. "base" points at a matrix
// whose ROWS are the N-columns with K contiguous (i.e. W[n][k] for C=A*W^T).
__device__ __forceinline__ v16bf load_bfrag(const bf16* base, int stride) {
    const int lane = threadIdx.x & 31;
    const bf16* p = base + (lane & 15) * stride + 16 * (lane >> 4);
    return cat8(*(const v8bf*)p, *(const v8bf*)(p + 8));
}

// ---------------------------------------------------------------------------
// fp32 -> bf16 conversion (weights)
// ---------------------------------------------------------------------------
__global__ __launch_bounds__(256)
void cvt_kernel(const float* __restrict__ in, bf16* __restrict__ out, int n) {
    int i = blockIdx.x * 256 + threadIdx.x;
    if (i < n) out[i] = f2bf(in[i]);
}

// ---------------------------------------------------------------------------
// LayerNorm: one block (256 thr) per row of 1024, bf16 output
// ---------------------------------------------------------------------------
__global__ __launch_bounds__(256)
void ln_kernel(const float* __restrict__ X, const float* __restrict__ g,
               const float* __restrict__ bb, bf16* __restrict__ Y) {
    constexpr int D = 1024;
    __shared__ float red[8];
    const int row = blockIdx.x, tid = threadIdx.x;
    const int wave = tid >> 5, lane = tid & 31;
    const float* xr = X + (size_t)row * D;
    v4f v = *(const v4f*)(xr + tid * 4);
    float s = v[0] + v[1] + v[2] + v[3];
#pragma unroll
    for (int o = 1; o < 32; o <<= 1) s += __shfl_xor(s, o, 32);
    if (lane == 0) red[wave] = s;
    __syncthreads();
    float tot = red[0]+red[1]+red[2]+red[3]+red[4]+red[5]+red[6]+red[7];
    float mean = tot * (1.0f / D);
    __syncthreads();
    float d0 = v[0]-mean, d1 = v[1]-mean, d2 = v[2]-mean, d3 = v[3]-mean;
    s = d0*d0 + d1*d1 + d2*d2 + d3*d3;
#pragma unroll
    for (int o = 1; o < 32; o <<= 1) s += __shfl_xor(s, o, 32);
    if (lane == 0) red[wave] = s;
    __syncthreads();
    tot = red[0]+red[1]+red[2]+red[3]+red[4]+red[5]+red[6]+red[7];
    float rstd = rsqrtf(tot * (1.0f / D) + 1e-5f);
    const int c = tid * 4;
    v4f gw = *(const v4f*)(g + c);
    v4f gb = *(const v4f*)(bb + c);
    bf16 o4[4];
    o4[0] = f2bf(d0 * rstd * gw[0] + gb[0]);
    o4[1] = f2bf(d1 * rstd * gw[1] + gb[1]);
    o4[2] = f2bf(d2 * rstd * gw[2] + gb[2]);
    o4[3] = f2bf(d3 * rstd * gw[3] + gb[3]);
    uint64_t pack; __builtin_memcpy(&pack, o4, 8);
    *(uint64_t*)(Y + (size_t)row * D + c) = pack;
}

// ---------------------------------------------------------------------------
// Tiled WMMA GEMM:  C(MxN) = epi( A(MxK) @ W(NxK)^T )
// Block = 256 thr (8 waves), tile 128x128, K-step 32.
// Double-buffered LDS staging via CDNA5 async global->LDS DMA (ASYNCcnt).
// Waves: 2 (M) x 4 (N); each wave computes 64x32 = 4x2 WMMA tiles.
// ---------------------------------------------------------------------------
enum { EPI_BF16 = 0, EPI_BIAS_GELU_BF16 = 1, EPI_RES_F32 = 2, EPI_BIAS_RES_F32 = 3 };
#define LDSTR 40   // padded LDS row stride (elements): 80B rows, conflict-free

template<int EPI>
__global__ __launch_bounds__(256)
void gemm_kernel(const bf16* __restrict__ A, const bf16* __restrict__ W,
                 const float* __restrict__ bias, const float* __restrict__ resid,
                 void* __restrict__ Cout, int M, int N, int K) {
    __shared__ __align__(16) bf16 As[2][128 * LDSTR];
    __shared__ __align__(16) bf16 Bs[2][128 * LDSTR];
    const int tid  = threadIdx.x;
    const int wave = tid >> 5, lane = tid & 31;
    const int wr = wave >> 2, wc = wave & 3;           // 2 x 4 wave grid
    const int m0 = blockIdx.y * 128, n0 = blockIdx.x * 128;

    v8f acc[4][2];
#pragma unroll
    for (int i = 0; i < 4; ++i)
#pragma unroll
        for (int j = 0; j < 2; ++j)
#pragma unroll
            for (int r = 0; r < 8; ++r) acc[i][j][r] = 0.0f;

    // stage one 128x32 A tile + 128x32 W tile: 512 16B chunks each,
    // 2+2 async DMA ops per thread (4 per wave slot => ASYNCcnt +4 per tile)
    auto stage = [&](int buf, int kk) {
#pragma unroll
        for (int u = 0; u < 2; ++u) {
            const int c = tid + u * 256;
            const int row = c >> 2, col = (c & 3) * 8;
            async_copy16(&As[buf][row * LDSTR + col],
                         A + (size_t)(m0 + row) * K + kk + col);
            async_copy16(&Bs[buf][row * LDSTR + col],
                         W + (size_t)(n0 + row) * K + kk + col);
        }
    };

    const int steps = K >> 5;
    stage(0, 0);
    for (int it = 0; it < steps; ++it) {
        const int buf = it & 1;
        const bool more = (it + 1) < steps;
        if (more) stage(buf ^ 1, (it + 1) * 32);       // prefetch next tile
        wait_async_le(more ? 4 : 0);                   // tile `it` landed in LDS
        __syncthreads();
        v16bf af[4], bfx[2];
#pragma unroll
        for (int i = 0; i < 4; ++i) af[i]  = load_afrag(&As[buf][(wr*64 + 16*i) * LDSTR], LDSTR);
#pragma unroll
        for (int j = 0; j < 2; ++j) bfx[j] = load_bfrag(&Bs[buf][(wc*32 + 16*j) * LDSTR], LDSTR);
#pragma unroll
        for (int i = 0; i < 4; ++i)
#pragma unroll
            for (int j = 0; j < 2; ++j)
                acc[i][j] = wmma_bf16(af[i], bfx[j], acc[i][j]);
        __syncthreads();                               // protect buf before rewrite
    }

    // epilogue: C-tile layout: lane n = lane&15 is column; VGPR r -> row r + 8*(lane>>4)
    const int hf = lane >> 4, ln = lane & 15;
#pragma unroll
    for (int i = 0; i < 4; ++i)
#pragma unroll
        for (int j = 0; j < 2; ++j) {
            const int col = n0 + wc * 32 + 16 * j + ln;
            float bval = 0.0f;
            if constexpr (EPI == EPI_BIAS_GELU_BF16 || EPI == EPI_BIAS_RES_F32)
                bval = bias[col];
#pragma unroll
            for (int r = 0; r < 8; ++r) {
                const int row = m0 + wr * 64 + 16 * i + r + 8 * hf;
                const size_t idx = (size_t)row * N + col;
                float v = acc[i][j][r];
                if constexpr (EPI == EPI_BF16) {
                    ((bf16*)Cout)[idx] = f2bf(v);
                } else if constexpr (EPI == EPI_BIAS_GELU_BF16) {
                    v += bval;
                    v = 0.5f * v * (1.0f + erff(v * 0.70710678118654752f));
                    ((bf16*)Cout)[idx] = f2bf(v);
                } else if constexpr (EPI == EPI_RES_F32) {
                    ((float*)Cout)[idx] = resid[idx] + v;
                } else {
                    ((float*)Cout)[idx] = resid[idx] + v + bval;
                }
            }
        }
}

// ---------------------------------------------------------------------------
// RoPE: qkv (B*L, 3*1024) bf16 -> Q,K (bh, L, 64) bf16 and V^T (bh, 64, L) bf16
// ---------------------------------------------------------------------------
__global__ __launch_bounds__(256)
void rope_kernel(const bf16* __restrict__ qkv, const float* __restrict__ cs,
                 const float* __restrict__ sn, bf16* __restrict__ Q,
                 bf16* __restrict__ Ko, bf16* __restrict__ Vt) {
    const int idx = blockIdx.x * 256 + threadIdx.x;   // ((b*16+h)*2048 + l)*64 + d
    const int d  = idx & 63;
    const int l  = (idx >> 6) & 2047;
    const int bh = idx >> 17;
    const int b = bh >> 4, h = bh & 15;
    const size_t base = (size_t)(b * 2048 + l) * 3072;
    const float c = cs[l * 64 + d];
    const float s = sn[l * 64 + d];
    const float qd = bf2f(qkv[base +        h * 64 + d]);
    const float qp = bf2f(qkv[base +        h * 64 + (d ^ 1)]);
    const float kd = bf2f(qkv[base + 1024 + h * 64 + d]);
    const float kp = bf2f(qkv[base + 1024 + h * 64 + (d ^ 1)]);
    const float qrot = (d & 1) ? qp : -qp;
    const float krot = (d & 1) ? kp : -kp;
    Q [((size_t)bh * 2048 + l) * 64 + d] = f2bf(qd * c + qrot * s);
    Ko[((size_t)bh * 2048 + l) * 64 + d] = f2bf(kd * c + krot * s);
    Vt[((size_t)bh * 64 + d) * 2048 + l] = qkv[base + 2048 + h * 64 + d];
}

// ---------------------------------------------------------------------------
// Flash attention: grid (B*H, L/64), block 128 (4 waves), wave = 16 q-rows.
// S = QK^T via WMMA, online softmax (rows live across 16-lane halves of the
// C layout), P re-laid-out through a per-wave LDS tile into the A-fragment
// layout, O += P @ V via WMMA with V pre-transposed (Dh x L).
// ---------------------------------------------------------------------------
__global__ __launch_bounds__(128)
void attn_kernel(const bf16* __restrict__ Q, const bf16* __restrict__ Km,
                 const bf16* __restrict__ Vt, bf16* __restrict__ Out) {
    constexpr int L = 2048, Dh = 64;
    __shared__ __align__(16) bf16 P[4][16 * LDSTR];
    const int bh = blockIdx.x;
    const int b = bh >> 4, h = bh & 15;
    const int wave = threadIdx.x >> 5, lane = threadIdx.x & 31;
    const int hf = lane >> 4, ln = lane & 15;
    const int q0 = blockIdx.y * 64 + wave * 16;

    const bf16* Qb = Q  + ((size_t)bh * L + q0) * Dh;
    const bf16* Kb = Km + (size_t)bh * L * Dh;
    const bf16* Vb = Vt + (size_t)bh * Dh * L;

    v16bf aq[2];
#pragma unroll
    for (int f = 0; f < 2; ++f) {
        const bf16* p = Qb + ln * Dh + f * 32 + 16 * hf;
        aq[f] = cat8(*(const v8bf*)p, *(const v8bf*)(p + 8));
    }

    float mi[8], li[8];
    v8f o[4];
#pragma unroll
    for (int r = 0; r < 8; ++r) { mi[r] = -1e30f; li[r] = 0.0f; }
#pragma unroll
    for (int n = 0; n < 4; ++n)
#pragma unroll
        for (int r = 0; r < 8; ++r) o[n][r] = 0.0f;

    const float scale = 0.125f;  // 1/sqrt(64)
    for (int kt = 0; kt < L; kt += 32) {
        // speculative prefetch of next K/V tiles into the cache hierarchy
        if (kt + 32 < L) {
            __builtin_prefetch(Kb + (size_t)(kt + 32 + ln) * Dh, 0, 1);
            __builtin_prefetch(Vb + (size_t)ln * L + kt + 32, 0, 1);
        }
        v8f s[2];
#pragma unroll
        for (int t = 0; t < 2; ++t) {
            const bf16* krow = Kb + (size_t)(kt + t * 16 + ln) * Dh;
            v16bf kb0 = cat8(*(const v8bf*)(krow + 16 * hf),
                             *(const v8bf*)(krow + 16 * hf + 8));
            v16bf kb1 = cat8(*(const v8bf*)(krow + 32 + 16 * hf),
                             *(const v8bf*)(krow + 32 + 16 * hf + 8));
            v8f c;
#pragma unroll
            for (int r = 0; r < 8; ++r) c[r] = 0.0f;
            c = wmma_bf16(aq[0], kb0, c);
            c = wmma_bf16(aq[1], kb1, c);
            s[t] = c;
        }
        float alpha[8];
#pragma unroll
        for (int r = 0; r < 8; ++r) {
            float v0 = s[0][r] * scale, v1 = s[1][r] * scale;
            float mx = fmaxf(v0, v1);
#pragma unroll
            for (int off = 1; off < 16; off <<= 1) mx = fmaxf(mx, __shfl_xor(mx, off, 16));
            const float mn = fmaxf(mi[r], mx);
            const float al = __expf(mi[r] - mn);
            const float p0 = __expf(v0 - mn), p1 = __expf(v1 - mn);
            float rs = p0 + p1;
#pragma unroll
            for (int off = 1; off < 16; off <<= 1) rs += __shfl_xor(rs, off, 16);
            li[r] = li[r] * al + rs;
            mi[r] = mn;
            alpha[r] = al;
            P[wave][(r + 8 * hf) * LDSTR + ln]      = f2bf(p0);
            P[wave][(r + 8 * hf) * LDSTR + 16 + ln] = f2bf(p1);
        }
#pragma unroll
        for (int n = 0; n < 4; ++n)
#pragma unroll
            for (int r = 0; r < 8; ++r) o[n][r] *= alpha[r];
        __syncthreads();
        v16bf pa = load_afrag(&P[wave][0], LDSTR);
#pragma unroll
        for (int n = 0; n < 4; ++n) {
            const bf16* vrow = Vb + (size_t)(n * 16 + ln) * L + kt + 16 * hf;
            v16bf vb = cat8(*(const v8bf*)vrow, *(const v8bf*)(vrow + 8));
            o[n] = wmma_bf16(pa, vb, o[n]);
        }
        __syncthreads();
    }
#pragma unroll
    for (int n = 0; n < 4; ++n)
#pragma unroll
        for (int r = 0; r < 8; ++r) {
            const int row = q0 + r + 8 * hf;
            const float v = o[n][r] / li[r];
            Out[((size_t)b * L + row) * 1024 + h * 64 + n * 16 + ln] = f2bf(v);
        }
}

// ---------------------------------------------------------------------------
// Host-side orchestration
// ---------------------------------------------------------------------------
extern "C" void kernel_launch(void* const* d_in, const int* in_sizes, int n_in,
                              void* d_out, int out_size, void* d_ws, size_t ws_size,
                              hipStream_t stream) {
    const float* x      = (const float*)d_in[0];
    const float* cosb   = (const float*)d_in[1];
    const float* sinb   = (const float*)d_in[2];
    const float* ln1_w  = (const float*)d_in[3];
    const float* ln1_b  = (const float*)d_in[4];
    const float* qkv_w  = (const float*)d_in[5];
    const float* proj_w = (const float*)d_in[6];
    const float* ln2_w  = (const float*)d_in[7];
    const float* ln2_b  = (const float*)d_in[8];
    const float* ff1_w  = (const float*)d_in[9];
    const float* ff1_b  = (const float*)d_in[10];
    const float* ff2_w  = (const float*)d_in[11];
    const float* ff2_b  = (const float*)d_in[12];

    constexpr int B = 2, L = 2048, D = 1024, H = 16, Dh = 64;
    constexpr int M = B * L;                     // 4096 token rows

    char* ws = (char*)d_ws;
    size_t off = 0;
    auto take = [&](size_t bytes) -> char* {
        char* p = ws + off;
        off = (off + bytes + 255) & ~(size_t)255;
        return p;
    };
    bf16* wq    = (bf16*)take((size_t)3 * D * D * 2);
    bf16* wp    = (bf16*)take((size_t)D * D * 2);
    bf16* wf1   = (bf16*)take((size_t)4 * D * D * 2);
    bf16* wf2   = (bf16*)take((size_t)4 * D * D * 2);
    bf16* h1    = (bf16*)take((size_t)M * D * 2);
    bf16* qkvb  = (bf16*)take((size_t)M * 3 * D * 2);
    bf16* qb    = (bf16*)take((size_t)B * H * L * Dh * 2);
    bf16* kb    = (bf16*)take((size_t)B * H * L * Dh * 2);
    bf16* vtb   = (bf16*)take((size_t)B * H * Dh * L * 2);
    bf16* attnb = (bf16*)take((size_t)M * D * 2);
    float* x1   = (float*)take((size_t)M * D * 4);
    bf16* h2    = (bf16*)take((size_t)M * D * 2);
    bf16* ffact = (bf16*)take((size_t)M * 4 * D * 2);

    // weights -> bf16
    cvt_kernel<<<(3 * D * D + 255) / 256, 256, 0, stream>>>(qkv_w,  wq,  3 * D * D);
    cvt_kernel<<<(D * D + 255) / 256,     256, 0, stream>>>(proj_w, wp,  D * D);
    cvt_kernel<<<(4 * D * D + 255) / 256, 256, 0, stream>>>(ff1_w,  wf1, 4 * D * D);
    cvt_kernel<<<(4 * D * D + 255) / 256, 256, 0, stream>>>(ff2_w,  wf2, 4 * D * D);

    // LN1
    ln_kernel<<<M, 256, 0, stream>>>(x, ln1_w, ln1_b, h1);
    // QKV = h1 @ qkv_w^T           (4096 x 3072 x 1024)
    gemm_kernel<EPI_BF16><<<dim3(3 * D / 128, M / 128), 256, 0, stream>>>(
        h1, wq, nullptr, nullptr, qkvb, M, 3 * D, D);
    // RoPE + head split + V transpose
    rope_kernel<<<(B * H * L * Dh) / 256, 256, 0, stream>>>(qkvb, cosb, sinb, qb, kb, vtb);
    // attention
    attn_kernel<<<dim3(B * H, L / 64), 128, 0, stream>>>(qb, kb, vtb, attnb);
    // x1 = x + attn @ proj_w^T     (4096 x 1024 x 1024)
    gemm_kernel<EPI_RES_F32><<<dim3(D / 128, M / 128), 256, 0, stream>>>(
        attnb, wp, nullptr, x, x1, M, D, D);
    // LN2
    ln_kernel<<<M, 256, 0, stream>>>(x1, ln2_w, ln2_b, h2);
    // ff = gelu(h2 @ ff1_w^T + b)  (4096 x 4096 x 1024)
    gemm_kernel<EPI_BIAS_GELU_BF16><<<dim3(4 * D / 128, M / 128), 256, 0, stream>>>(
        h2, wf1, ff1_b, nullptr, ffact, M, 4 * D, D);
    // out = x1 + ff @ ff2_w^T + b  (4096 x 1024 x 4096)
    gemm_kernel<EPI_BIAS_RES_F32><<<dim3(D / 128, M / 128), 256, 0, stream>>>(
        ffact, wf2, ff2_b, x1, (float*)d_out, M, D, 4 * D);
}
